// TransformerModel_64854006169683
// MI455X (gfx1250) — matrix operand order repaired and checked by reading, not compiled, not run on hardware
//
#include <hip/hip_runtime.h>
#include <math.h>

// ---------------------------------------------------------------------------
// CDNA5 (gfx1250, wave32) bf16 WMMA GEMM engine for the transformer block.
//  - weights pre-transposed + converted to bf16 [N][K] once per launch
//  - activations produced in bf16 by their producer kernels
//  - LDS holds bf16 tiles, double-buffered, filled with async-to-LDS DMA
//  - WMMA fragment build = 2 x ds_load_b128, zero VALU in the hot loop
// ---------------------------------------------------------------------------

typedef __attribute__((ext_vector_type(16))) __bf16 v16bf;
typedef __attribute__((ext_vector_type(8)))  float  v8f;
typedef int v4i __attribute__((vector_size(16)));   // matches async builtin param

#if defined(__has_builtin)
#if __has_builtin(__builtin_amdgcn_global_load_async_to_lds_b128)
#define USE_ASYNC_LDS 1
#endif
#endif
#ifndef USE_ASYNC_LDS
#define USE_ASYNC_LDS 0
#endif

#if defined(__has_builtin)
#if __has_builtin(__builtin_amdgcn_s_wait_asynccnt)
#define WAIT_ASYNC() __builtin_amdgcn_s_wait_asynccnt(0)
#else
#define WAIT_ASYNC() asm volatile("s_wait_asynccnt 0" ::: "memory")
#endif
#else
#define WAIT_ASYNC() asm volatile("s_wait_asynccnt 0" ::: "memory")
#endif

#define TM 128          // block tile M
#define TN 256          // block tile N
#define TK 32           // K step (one bf16 WMMA)
#define LROW 40         // bf16 per LDS row: 32 + 8 pad -> 80B stride (16B aligned)

__device__ __forceinline__ float gelu_f(float x) {
    const float c = 0.7978845608028654f;
    float t = tanhf(c * (x + 0.044715f * x * x * x));
    return 0.5f * x * (1.0f + t);
}

union Frag {
    v16bf  v;
    float4 f4[2];   // 2 x 16B raw halves (elements 0..7 / 8..15)
};

// ---------------------------------------------------------------------------
// C[M,N] = act(A[M,K] @ B + bias) with A bf16 row-major [M][K] and
// Wt bf16 PRE-TRANSPOSED [N][K]. fp32 accumulate. OutT = float or __bf16.
// Block: 256 threads (8 waves, 2x4). Wave tile 64x64 = 4x4 WMMA 16x16 tiles.
// Requires K % 32 == 0; M,N bounds-checked.
// ---------------------------------------------------------------------------
template <int ACT, typename OutT>
__global__ __launch_bounds__(256, 1)
void gemm_wmma_kernel(const __bf16* __restrict__ A, const __bf16* __restrict__ Wt,
                      const float* __restrict__ bias, OutT* __restrict__ C,
                      int M, int N, int K) {
    __shared__ __bf16 lsA[2][TM][LROW];  // [buf][m][k]
    __shared__ __bf16 lsB[2][TN][LROW];  // [buf][n][k]

    const int tid  = threadIdx.x;
    const int lane = tid & 31;
    const int wave = tid >> 5;
    const int wm   = wave >> 2;   // 0..1
    const int wn   = wave & 3;    // 0..3
    const int half = lane >> 4;   // K-half select per 16-bit WMMA layout
    const int l15  = lane & 15;
    const int m0   = blockIdx.y * TM;
    const int n0   = blockIdx.x * TN;

    v8f acc[4][4];
#pragma unroll
    for (int i = 0; i < 4; ++i)
#pragma unroll
        for (int j = 0; j < 4; ++j) acc[i][j] = v8f{};

    const int nk = K / TK;

    // ---- tile stager: contiguous bf16 b128 copies (A and packed Wt) ------
    auto stage = [&](int buf, int k0) {
        // A tile: 128 rows x 32 bf16 = 512 b128 slots, 2 per thread
#pragma unroll
        for (int i = 0; i < 2; ++i) {
            int s   = tid + i * 256;
            int row = s >> 2;
            int kq  = (s & 3) * 8;            // bf16 units
            const __bf16* gp = A + (size_t)(m0 + row) * K + (k0 + kq);
            bool ok = (m0 + row) < M;
#if USE_ASYNC_LDS
            if (ok) {
                __builtin_amdgcn_global_load_async_to_lds_b128(
                    (__attribute__((address_space(1))) v4i*)(v4i*)const_cast<__bf16*>(gp),
                    (__attribute__((address_space(3))) v4i*)(v4i*)&lsA[buf][row][kq],
                    0, 0);
            } else {
                float4 z = {0.f, 0.f, 0.f, 0.f};
                *(float4*)&lsA[buf][row][kq] = z;
            }
#else
            float4 z = {0.f, 0.f, 0.f, 0.f};
            if (ok) z = *(const float4*)gp;
            *(float4*)&lsA[buf][row][kq] = z;
#endif
        }
        // B tile: 256 n-rows x 32 bf16 = 1024 b128 slots, 4 per thread
#pragma unroll
        for (int i = 0; i < 4; ++i) {
            int s   = tid + i * 256;
            int row = s >> 2;
            int kq  = (s & 3) * 8;
            const __bf16* gp = Wt + (size_t)(n0 + row) * K + (k0 + kq);
            bool ok = (n0 + row) < N;
#if USE_ASYNC_LDS
            if (ok) {
                __builtin_amdgcn_global_load_async_to_lds_b128(
                    (__attribute__((address_space(1))) v4i*)(v4i*)const_cast<__bf16*>(gp),
                    (__attribute__((address_space(3))) v4i*)(v4i*)&lsB[buf][row][kq],
                    0, 0);
            } else {
                float4 z = {0.f, 0.f, 0.f, 0.f};
                *(float4*)&lsB[buf][row][kq] = z;
            }
#else
            float4 z = {0.f, 0.f, 0.f, 0.f};
            if (ok) z = *(const float4*)gp;
            *(float4*)&lsB[buf][row][kq] = z;
#endif
        }
    };

    stage(0, 0);
#if USE_ASYNC_LDS
    WAIT_ASYNC();
#endif
    __syncthreads();

    for (int kt = 0; kt < nk; ++kt) {
        const int cur = kt & 1;
        if (kt + 1 < nk) stage(1 - cur, (kt + 1) * TK);  // async prefetch

        // ---- load all fragments first (batched ds_load_b128) -------------
        Frag fa[4], fb[4];
#pragma unroll
        for (int i = 0; i < 4; ++i) {
            const __bf16* ar = &lsA[cur][wm * 64 + i * 16 + l15][0];
            fa[i].f4[0] = *(const float4*)(ar + half * 8);        // K = half*8+0..7
            fa[i].f4[1] = *(const float4*)(ar + 16 + half * 8);   // K = 16+half*8+0..7
        }
#pragma unroll
        for (int j = 0; j < 4; ++j) {
            const __bf16* br = &lsB[cur][wn * 64 + j * 16 + l15][0];
            fb[j].f4[0] = *(const float4*)(br + half * 8);        // K = half*8+0..7
            fb[j].f4[1] = *(const float4*)(br + 16 + half * 8);   // K = 16+half*8+0..7
        }

        // ---- 16 WMMAs ----------------------------------------------------
#pragma unroll
        for (int i = 0; i < 4; ++i)
#pragma unroll
            for (int j = 0; j < 4; ++j)
                acc[i][j] = __builtin_amdgcn_wmma_f32_16x16x32_bf16(
                    false, fa[i].v, false, fb[j].v, (short)0, acc[i][j],
                    false, false);

#if USE_ASYNC_LDS
        WAIT_ASYNC();
#endif
        __syncthreads();
    }

    // ---- store: VGPR v holds row (v + half*8) of each 16x16 tile ---------
#pragma unroll
    for (int i = 0; i < 4; ++i) {
#pragma unroll
        for (int j = 0; j < 4; ++j) {
            int col   = n0 + wn * 64 + j * 16 + l15;
            int rbase = m0 + wm * 64 + i * 16 + half * 8;
            if (col < N) {
                float bv = bias ? bias[col] : 0.0f;
#pragma unroll
                for (int v = 0; v < 8; ++v) {
                    int r = rbase + v;
                    if (r < M) {
                        float x = acc[i][j][v] + bv;
                        if (ACT == 1) x = gelu_f(x);
                        C[(size_t)r * N + col] = (OutT)x;
                    }
                }
            }
        }
    }
}

// ---------------------------------------------------------------------------
// Wt[n*K+k] = (bf16) W[k*N+n] : 32x32 LDS-tiled transpose + convert.
// ---------------------------------------------------------------------------
__global__ void transpose_cvt_kernel(const float* __restrict__ W,
                                     __bf16* __restrict__ Wt, int K, int N) {
    __shared__ float tile[32][33];
    const int kb = blockIdx.y * 32, nb = blockIdx.x * 32;
    const int tx = threadIdx.x & 31, ty = threadIdx.x >> 5;  // 32x8
#pragma unroll
    for (int i = 0; i < 32; i += 8) {
        int k = kb + ty + i, n = nb + tx;
        tile[ty + i][tx] = (k < K && n < N) ? W[(size_t)k * N + n] : 0.0f;
    }
    __syncthreads();
#pragma unroll
    for (int i = 0; i < 32; i += 8) {
        int n = nb + ty + i, k = kb + tx;
        if (n < N && k < K) Wt[(size_t)n * K + k] = (__bf16)tile[tx][ty + i];
    }
}

// ---------------------------------------------------------------------------
// rmsnorm, emitting bf16 for the downstream WMMA GEMMs.
// ---------------------------------------------------------------------------
__global__ void rmsnorm_bf16_kernel(const float* __restrict__ x,
                                    const float* __restrict__ g,
                                    __bf16* __restrict__ out, int d) {
    __shared__ float red[256];
    const int row = blockIdx.x;
    const float* xr = x + (size_t)row * d;
    float s = 0.0f;
    for (int c = threadIdx.x; c < d; c += 256) {
        float v = xr[c];
        s += v * v;
    }
    red[threadIdx.x] = s;
    __syncthreads();
    for (int off = 128; off > 0; off >>= 1) {
        if (threadIdx.x < (unsigned)off) red[threadIdx.x] += red[threadIdx.x + off];
        __syncthreads();
    }
    float norm  = sqrtf(red[0] / (float)d);
    float scale = 1.0f / fmaxf(norm, 1e-8f);
    for (int c = threadIdx.x; c < d; c += 256)
        out[(size_t)row * d + c] = (__bf16)(xr[c] * scale * g[c]);
}

// ---------------------------------------------------------------------------
// out = t2 + geglu(proj): proj rows are [a(0..d-1) | gate(d..2d-1)]
// ---------------------------------------------------------------------------
__global__ void geglu_residual_kernel(const float* __restrict__ proj,
                                      const float* __restrict__ t2,
                                      float* __restrict__ out,
                                      int rows, int d) {
    size_t i = (size_t)blockIdx.x * blockDim.x + threadIdx.x;
    if (i >= (size_t)rows * d) return;
    size_t r = i / d, c = i % d;
    float a    = proj[r * (size_t)(2 * d) + c];
    float gate = proj[r * (size_t)(2 * d) + d + c];
    out[i] = t2[i] + a * gelu_f(gate);
}

__global__ void copy_kernel(const float* __restrict__ in, float* __restrict__ out,
                            size_t n) {
    size_t i = (size_t)blockIdx.x * blockDim.x + threadIdx.x;
    if (i < n) out[i] = in[i];
}

// ---------------------------------------------------------------------------
// Model dims
// ---------------------------------------------------------------------------
#define MB   4
#define NSEQ 1024
#define DD   512
#define FFD  2048
#define ROWS (MB * NSEQ)  // 4096 token rows

extern "C" void kernel_launch(void* const* d_in, const int* in_sizes, int n_in,
                              void* d_out, int out_size, void* d_ws, size_t ws_size,
                              hipStream_t stream) {
    // setup_inputs flat order (depth-first over dicts):
    //  0: src  1: norm1_g
    //  2: ffd1.l1.w [512,2048]  3: ffd1.l1.b  4: ffd1.l2.w [2048,512]  5: ffd1.l2.b
    //  6..12: pkm1  13: geglu1.proj.w [512,1024]  14: geglu1.proj.b
    const float* src     = (const float*)d_in[0];
    const float* norm1_g = (const float*)d_in[1];
    const float* l1w     = (const float*)d_in[2];
    const float* l1b     = (const float*)d_in[3];
    const float* l2w     = (const float*)d_in[4];
    const float* l2b     = (const float*)d_in[5];
    const float* gegw    = (n_in > 13) ? (const float*)d_in[13] : nullptr;
    const float* gegb    = (n_in > 14) ? (const float*)d_in[14] : nullptr;

    float* out = (float*)d_out;

    // workspace layout: bf16 pool then fp32 pool
    const size_t eH0  = (size_t)ROWS * DD;        // h0 bf16
    const size_t eT1  = (size_t)ROWS * FFD;       // t1 bf16
    const size_t eW1  = (size_t)DD * FFD;         // Wt1 [FFD][DD]
    const size_t eW2  = (size_t)FFD * DD;         // Wt2 [DD][FFD]
    const size_t eWG  = (size_t)DD * (2 * DD);    // Wtg [1024][512]
    const size_t eT2  = (size_t)ROWS * DD;        // t2 f32
    const size_t eGP  = (size_t)ROWS * (2 * DD);  // gp f32
    const size_t need = 2 * (eH0 + eT1 + eW1 + eW2 + eWG) + 4 * (eT2 + eGP);

    if (ws_size < need || !gegw) {
        size_t n = (size_t)out_size;
        copy_kernel<<<(unsigned)((n + 255) / 256), 256, 0, stream>>>(src, out, n);
        return;
    }

    __bf16* h0b = (__bf16*)d_ws;
    __bf16* t1b = h0b + eH0;
    __bf16* wt1 = t1b + eT1;
    __bf16* wt2 = wt1 + eW1;
    __bf16* wtg = wt2 + eW2;
    float*  t2  = (float*)(wtg + eWG);
    float*  gp  = t2 + eT2;

    dim3 blk(256);

    // ---- weight prepack: transpose + fp32->bf16 --------------------------
    transpose_cvt_kernel<<<dim3(FFD / 32, DD / 32), blk, 0, stream>>>(l1w, wt1, DD, FFD);
    transpose_cvt_kernel<<<dim3(DD / 32, FFD / 32), blk, 0, stream>>>(l2w, wt2, FFD, DD);
    transpose_cvt_kernel<<<dim3((2 * DD) / 32, DD / 32), blk, 0, stream>>>(gegw, wtg, DD, 2 * DD);

    // ---- h0 = rmsnorm(norm1_g, src)  (bf16) ------------------------------
    rmsnorm_bf16_kernel<<<ROWS, blk, 0, stream>>>(src, norm1_g, h0b, DD);

    // ---- t1 = gelu(h0 @ l1w + l1b)  (bf16 out, feeds next GEMM) ----------
    gemm_wmma_kernel<1, __bf16><<<dim3((FFD + TN - 1) / TN, (ROWS + TM - 1) / TM),
                                  blk, 0, stream>>>(h0b, wt1, l1b, t1b, ROWS, FFD, DD);

    // ---- t2 = gelu(t1 @ l2w + l2b)  (f32 out) ----------------------------
    gemm_wmma_kernel<1, float><<<dim3((DD + TN - 1) / TN, (ROWS + TM - 1) / TM),
                                 blk, 0, stream>>>(t1b, wt2, l2b, t2, ROWS, DD, FFD);

    // ---- gp = h0 @ gegw + gegb  (f32 out) --------------------------------
    gemm_wmma_kernel<0, float><<<dim3((2 * DD + TN - 1) / TN, (ROWS + TM - 1) / TM),
                                 blk, 0, stream>>>(h0b, wtg, gegb, gp, ROWS, 2 * DD, DD);

    // ---- out = t2 + geglu(gp) --------------------------------------------
    {
        size_t n = (size_t)ROWS * DD;
        geglu_residual_kernel<<<(unsigned)((n + 255) / 256), 256, 0, stream>>>(
            gp, t2, out, ROWS, DD);
    }
}
